// SpectalRecurrentAttention_78288663872408
// MI455X (gfx1250) — matrix operand order repaired and checked by reading, not compiled, not run on hardware
//
#include <hip/hip_runtime.h>
#include <hip/hip_bf16.h>
#include <math.h>

#define T_STEPS 500
#define BATCH   16
#define FREQ    32
#define NROWS   (BATCH*FREQ)   // 512
#define CIN     64
#define HID     64

typedef __attribute__((ext_vector_type(16))) _Float16 v16h;
typedef __attribute__((ext_vector_type(8)))  float    v8f;

__device__ __forceinline__ v8f wmma_f16(v16h a, v16h b, v8f c) {
  // V_WMMA_F32_16X16X32_F16: D = A(16x32) * B(32x16) + C(16x16 f32)
  return __builtin_amdgcn_wmma_f32_16x16x32_f16(false, a, false, b, (short)0, c,
                                                false, false);
}

// Branchless transcendentals: v_exp_f32 + v_rcp_f32 only. Keeps EXEC uniform
// (WMMA requires EXEC all-1s) and avoids the IEEE-divide / libm-tanh blowup.
__device__ __forceinline__ float fast_sigmoid(float x) {
  return __builtin_amdgcn_rcpf(1.f + __expf(-x));   // saturates correctly
}
__device__ __forceinline__ float fast_tanh(float x) {
  return 1.f - 2.f * __builtin_amdgcn_rcpf(1.f + __expf(2.f * x));
}

// ---- CDNA5 async global->LDS copy (ASYNCcnt path, no VGPR round-trip) ----
// LDS operand is the low 32 bits of the generic address (LDS aperture).
__device__ __forceinline__ void async_b128(unsigned lds, const void* g) {
  asm volatile("global_load_async_to_lds_b128 %0, %1, off"
               :: "v"(lds), "v"((unsigned long long)g) : "memory");
}
__device__ __forceinline__ unsigned lds_addr_of(const void* p) {
  return (unsigned)(unsigned long long)p;
}

// ---- fragment builders (contiguous k = 16h + s, same order in A and B, wave32) ----

// A fragment: 16x32 f16, row-major f16 source, ld = row stride in halves.
// lanes 0-15: row=lane,   K = 0..15
// lanes 16-31: row=lane-16, K = 16..31
__device__ __forceinline__ v16h frag_a(const _Float16* src, int ld, unsigned lane) {
  unsigned row = lane & 15u;
  unsigned kb  = (lane & 16u) ? 16u : 0u;
  v16h a;
#pragma unroll
  for (int i = 0; i < 16; ++i) a[i] = src[row * ld + kb + i];
  return a;
}

// A fragment built from an f32 tile in LDS (async copy leaves x in f32);
// the f32->f16 cvt dual-issues with surrounding WMMA latency.
__device__ __forceinline__ v16h frag_a_f32(const float* src, int ld, unsigned lane) {
  unsigned row = lane & 15u;
  unsigned kb  = (lane & 16u) ? 16u : 0u;
  v16h a;
#pragma unroll
  for (int i = 0; i < 16; ++i) a[i] = (_Float16)src[row * ld + kb + i];
  return a;
}

// B fragment: 32x16 f16 (K x N) row-major, ld = row stride.
// lanes 0-15 hold K=0..15 (col = lane), lanes 16-31 hold K=16..31.
__device__ __forceinline__ v16h frag_b(const _Float16* src, int ld, unsigned lane) {
  unsigned n  = lane & 15u;
  unsigned kb = (lane & 16u) ? 16u : 0u;
  v16h b;
#pragma unroll
  for (int i = 0; i < 16; ++i) b[i] = src[(kb + i) * ld + n];
  return b;
}

// B = M^T chunk where M is N x K row-major f16:  b[k][n] = src[n*ld + k]
__device__ __forceinline__ v16h frag_bT(const _Float16* src, int ld, unsigned lane) {
  unsigned n  = lane & 15u;
  unsigned kb = (lane & 16u) ? 16u : 0u;
  v16h b;
#pragma unroll
  for (int i = 0; i < 16; ++i) b[i] = src[n * ld + kb + i];
  return b;
}

// B = W^T chunk where W is N x K row-major f32 (PyTorch weight layout):
// b[k][n] = (f16) W[(nbase+n)*ld + kofs + k]
__device__ __forceinline__ v16h frag_bT_f32(const float* w, int ld, int nbase,
                                            int kofs, unsigned lane) {
  unsigned n  = lane & 15u;
  unsigned kb = (lane & 16u) ? 16u : 0u;
  v16h b;
#pragma unroll
  for (int i = 0; i < 16; ++i)
    b[i] = (_Float16)w[(size_t)(nbase + n) * ld + kofs + kb + i];
  return b;
}

// =====================================================================
// Kernel 1: persistent 3-GRU recurrence.
// grid = 3 GRUs x 32 row-blocks(16 rows) = 96 WGs; block = 128 thr (4 waves).
// Each workgroup is self-contained (batch rows independent in a GRU).
// Wave w owns hidden columns [16w,16w+16) of all three gates.
// x tiles are double-buffered via GLOBAL_LOAD_ASYNC_TO_LDS_B128: tile t+1
// streams into LDS while tile t's WMMAs run -> HBM latency off the chain.
// =====================================================================
__global__ __launch_bounds__(128)
void gru3_persistent(const float* __restrict__ x,      // [T][512][64]
                     const float* __restrict__ state,  // [3][512][64]
                     const float* __restrict__ wih_q, const float* __restrict__ whh_q,
                     const float* __restrict__ bih_q, const float* __restrict__ bhh_q,
                     const float* __restrict__ wih_k, const float* __restrict__ whh_k,
                     const float* __restrict__ bih_k, const float* __restrict__ bhh_k,
                     const float* __restrict__ wih_v, const float* __restrict__ whh_v,
                     const float* __restrict__ bih_v, const float* __restrict__ bhh_v,
                     _Float16* __restrict__ qkv,       // [3][T][512][64] f16
                     float* __restrict__ state_out)    // [3][512][64]
{
  const int g    = blockIdx.x >> 5;        // 0..2  (q,k,v)
  const int mblk = blockIdx.x & 31;        // row block
  const int row0 = mblk * 16;

  const float* wih = (g == 0) ? wih_q : (g == 1) ? wih_k : wih_v;
  const float* whh = (g == 0) ? whh_q : (g == 1) ? whh_k : whh_v;
  const float* bih = (g == 0) ? bih_q : (g == 1) ? bih_k : bih_v;
  const float* bhh = (g == 0) ? bhh_q : (g == 1) ? bhh_k : bhh_v;

  const unsigned wave = threadIdx.x >> 5;
  const unsigned lane = threadIdx.x & 31u;
  const int hcol0 = wave * 16;
  const int col   = lane & 15;
  const int rbase = (lane & 16u) ? 8 : 0;

  __shared__ float    xsf[2][16][68];      // double-buffered x tile, raw f32
  __shared__ _Float16 hs[16][72];

  // ---- loop-invariant weight fragments (B = W^T chunks), K=64 -> 2 chunks ----
  v16h bxr0 = frag_bT_f32(wih, CIN,   0 + hcol0,  0, lane);
  v16h bxr1 = frag_bT_f32(wih, CIN,   0 + hcol0, 32, lane);
  v16h bxz0 = frag_bT_f32(wih, CIN,  64 + hcol0,  0, lane);
  v16h bxz1 = frag_bT_f32(wih, CIN,  64 + hcol0, 32, lane);
  v16h bxn0 = frag_bT_f32(wih, CIN, 128 + hcol0,  0, lane);
  v16h bxn1 = frag_bT_f32(wih, CIN, 128 + hcol0, 32, lane);
  v16h bhr0 = frag_bT_f32(whh, HID,   0 + hcol0,  0, lane);
  v16h bhr1 = frag_bT_f32(whh, HID,   0 + hcol0, 32, lane);
  v16h bhz0 = frag_bT_f32(whh, HID,  64 + hcol0,  0, lane);
  v16h bhz1 = frag_bT_f32(whh, HID,  64 + hcol0, 32, lane);
  v16h bhn0 = frag_bT_f32(whh, HID, 128 + hcol0,  0, lane);
  v16h bhn1 = frag_bT_f32(whh, HID, 128 + hcol0, 32, lane);

  const float biasr = bih[  0 + hcol0 + col] + bhh[  0 + hcol0 + col];
  const float biasz = bih[ 64 + hcol0 + col] + bhh[ 64 + hcol0 + col];
  const float bxnb  = bih[128 + hcol0 + col];
  const float bhnb  = bhh[128 + hcol0 + col];

  // ---- per-thread async-copy chunk: 32B (2 x b128) of the 16x64 f32 tile ----
  const int crow = threadIdx.x >> 3;        // 0..15 tile row (rows contiguous)
  const int ccol = (threadIdx.x & 7) * 8;   // float index within row
  const unsigned ldsx0 = lds_addr_of(&xsf[0][crow][ccol]);
  const unsigned ldsx1 = lds_addr_of(&xsf[1][crow][ccol]);
  const float* gx = x + ((size_t)row0 + crow) * CIN + ccol;

  // ---- init h (LDS f16 mirror + f32 register copy for exact state output) ----
  const float* st = state + (size_t)g * NROWS * HID + (size_t)row0 * HID;
  for (int e = threadIdx.x; e < 16 * HID; e += blockDim.x) {
    int r = e >> 6, c = e & 63;
    hs[r][c] = (_Float16)st[e];
  }
  float hreg[8];
#pragma unroll
  for (int i = 0; i < 8; ++i)
    hreg[i] = st[(size_t)(rbase + i) * HID + hcol0 + col];

  // prologue: stream tile 0 into buffer 0
  async_b128(ldsx0,      gx);
  async_b128(ldsx0 + 16, gx + 4);
  __syncthreads();

  for (int t = 0; t < T_STEPS; ++t) {
    const int buf = t & 1;
    if (t + 1 < T_STEPS) {
      // issue tile t+1 into the other buffer, then wait only for tile t
      // (async loads complete in order; <=2 outstanding == our new batch)
      const float* gp = gx + (size_t)(t + 1) * NROWS * CIN;
      async_b128(buf ? ldsx0 : ldsx1,        gp);
      async_b128((buf ? ldsx0 : ldsx1) + 16, gp + 4);
      asm volatile("s_wait_asynccnt 2" ::: "memory");
    } else {
      asm volatile("s_wait_asynccnt 0" ::: "memory");
    }
    __syncthreads();  // tile t in LDS everywhere; prev step's hs writes visible

    const float* xt = &xsf[buf][0][0];
    v16h ax0 = frag_a_f32(xt,      68, lane);
    v16h ax1 = frag_a_f32(xt + 32, 68, lane);
    v16h ah0 = frag_a(&hs[0][0],  72, lane);
    v16h ah1 = frag_a(&hs[0][32], 72, lane);

    v8f accr, accz, accxn, acchn;
#pragma unroll
    for (int i = 0; i < 8; ++i) {
      accr[i] = biasr; accz[i] = biasz; accxn[i] = bxnb; acchn[i] = bhnb;
    }
    // r,z: fuse x- and h-projections into one accumulator (bias preloaded)
    accr  = wmma_f16(ax0, bxr0, accr);  accr  = wmma_f16(ax1, bxr1, accr);
    accr  = wmma_f16(ah0, bhr0, accr);  accr  = wmma_f16(ah1, bhr1, accr);
    accz  = wmma_f16(ax0, bxz0, accz);  accz  = wmma_f16(ax1, bxz1, accz);
    accz  = wmma_f16(ah0, bhz0, accz);  accz  = wmma_f16(ah1, bhz1, accz);
    // n-gate needs x- and h-parts separate (r gates only the h-part)
    accxn = wmma_f16(ax0, bxn0, accxn); accxn = wmma_f16(ax1, bxn1, accxn);
    acchn = wmma_f16(ah0, bhn0, acchn); acchn = wmma_f16(ah1, bhn1, acchn);

#pragma unroll
    for (int i = 0; i < 8; ++i) {
      float r = fast_sigmoid(accr[i]);
      float z = fast_sigmoid(accz[i]);
      float n = fast_tanh(accxn[i] + r * acchn[i]);
      hreg[i] = n + z * (hreg[i] - n);     // (1-z)*n + z*h, one fewer FMA
    }
    __syncthreads();  // all waves done reading xsf/hs before overwrite

    _Float16* qv = qkv + (((size_t)g * T_STEPS + t) * NROWS + row0) * HID;
#pragma unroll
    for (int i = 0; i < 8; ++i) {
      _Float16 hv = (_Float16)hreg[i];
      hs[rbase + i][hcol0 + col] = hv;                       // next step's A
      qv[(size_t)(rbase + i) * HID + hcol0 + col] = hv;      // q/k/v stream
    }
  }

  // final hidden state -> d_out state section (f32, never round-tripped f16)
  float* so = state_out + (size_t)g * NROWS * HID + (size_t)row0 * HID;
#pragma unroll
  for (int i = 0; i < 8; ++i)
    so[(size_t)(rbase + i) * HID + hcol0 + col] = hreg[i];
}

// =====================================================================
// Kernel 2: per-(t,b) attention + freq-softmax + output projection.
// grid = 500*16 = 8000 WGs; block = 128 thr (4 waves).
// Q/K/V staged by pure async byte copies (already f16 in workspace).
// =====================================================================
__global__ __launch_bounds__(128)
void spectral_attention(const _Float16* __restrict__ qkv, // [3][T][512][64] f16
                        const float* __restrict__ w_o,    // [64][64]
                        const float* __restrict__ b_o,    // [64]
                        float* __restrict__ out)          // [T][16][32][64]
{
  const int t = blockIdx.x >> 4;
  const int b = blockIdx.x & 15;
  const unsigned wave = threadIdx.x >> 5;
  const unsigned lane = threadIdx.x & 31u;
  const int rbase = (lane & 16u) ? 8 : 0;
  const int cc    = lane & 15;

  __shared__ _Float16 Qs[32][72];
  __shared__ _Float16 Ks[32][72];
  __shared__ _Float16 Vs[32][72];
  __shared__ _Float16 Wt[32][40];   // attention weights Q@K^T (f16)
  __shared__ float    Af[32][68];   // pre-softmax W@V (f32)
  __shared__ _Float16 Ss[32][72];   // post-softmax (f16)

  const size_t plane = (size_t)T_STEPS * NROWS * HID;
  const size_t base  = ((size_t)t * NROWS + (size_t)b * FREQ) * HID;
  const _Float16* qg = qkv + base;
  const _Float16* kg = qkv + plane + base;
  const _Float16* vg = qkv + 2 * plane + base;

  // async-stage Q,K,V: per matrix 256 16B chunks; this thread does chunks
  // (tid, tid+128); chunk c -> row c>>3, halves (c&7)*8 (rows contiguous).
  {
    const int c0 = threadIdx.x, c1 = threadIdx.x + 128;
    async_b128(lds_addr_of(&Qs[c0 >> 3][(c0 & 7) * 8]), qg + c0 * 8);
    async_b128(lds_addr_of(&Qs[c1 >> 3][(c1 & 7) * 8]), qg + c1 * 8);
    async_b128(lds_addr_of(&Ks[c0 >> 3][(c0 & 7) * 8]), kg + c0 * 8);
    async_b128(lds_addr_of(&Ks[c1 >> 3][(c1 & 7) * 8]), kg + c1 * 8);
    async_b128(lds_addr_of(&Vs[c0 >> 3][(c0 & 7) * 8]), vg + c0 * 8);
    async_b128(lds_addr_of(&Vs[c1 >> 3][(c1 & 7) * 8]), vg + c1 * 8);
    asm volatile("s_wait_asynccnt 0" ::: "memory");
  }
  __syncthreads();

  // W = Q @ K^T  (32x32): 4 tiles, one per wave; K=64 -> 2 chunks
  {
    const int mt = (wave >> 1) & 1, nt = wave & 1;
    v8f acc = {};
#pragma unroll
    for (int kc = 0; kc < 64; kc += 32) {
      v16h a  = frag_a (&Qs[mt * 16][kc], 72, lane);
      v16h bb = frag_bT(&Ks[nt * 16][kc], 72, lane);   // B = K^T chunk
      acc = wmma_f16(a, bb, acc);
    }
#pragma unroll
    for (int i = 0; i < 8; ++i)
      Wt[mt * 16 + rbase + i][nt * 16 + cc] = (_Float16)acc[i];
  }
  __syncthreads();

  // A = W @ V  (32x64): K=32 -> single WMMA per tile; 8 tiles, 2 per wave
#pragma unroll
  for (int j = 0; j < 2; ++j) {
    const int tile = wave * 2 + j;
    const int mt = tile >> 2, nt = tile & 3;
    v8f acc = {};
    v16h a  = frag_a(&Wt[mt * 16][0], 40, lane);
    v16h bb = frag_b(&Vs[0][nt * 16], 72, lane);
    acc = wmma_f16(a, bb, acc);
#pragma unroll
    for (int i = 0; i < 8; ++i)
      Af[mt * 16 + rbase + i][nt * 16 + cc] = acc[i];
  }
  __syncthreads();

  // softmax over the 32 freq rows, per hidden column (axis=-2 in reference).
  // Each active thread owns one column exclusively; exp cached in Af.
  if (threadIdx.x < HID) {
    const int c = threadIdx.x;
    float m = -3.4e38f;
    for (int r = 0; r < FREQ; ++r) m = fmaxf(m, Af[r][c]);
    float s = 0.f;
    for (int r = 0; r < FREQ; ++r) {
      float e = __expf(Af[r][c] - m);
      Af[r][c] = e;
      s += e;
    }
    const float inv = __builtin_amdgcn_rcpf(s);
    for (int r = 0; r < FREQ; ++r)
      Ss[r][c] = (_Float16)(Af[r][c] * inv);
  }
  __syncthreads();

  // O = S @ w_o^T + b_o  (32x64): 8 tiles, 2 per wave; f32 result to HBM
  float* og = out + ((size_t)t * BATCH + b) * FREQ * CIN;
#pragma unroll
  for (int j = 0; j < 2; ++j) {
    const int tile = wave * 2 + j;
    const int mt = tile >> 2, nt = tile & 3;
    v8f acc;
    const float bias = b_o[nt * 16 + cc];
#pragma unroll
    for (int i = 0; i < 8; ++i) acc[i] = bias;
#pragma unroll
    for (int kc = 0; kc < 64; kc += 32) {
      v16h a  = frag_a(&Ss[mt * 16][kc], 72, lane);
      v16h bb = frag_bT_f32(w_o, HID, nt * 16, kc, lane);
      acc = wmma_f16(a, bb, acc);
    }
#pragma unroll
    for (int i = 0; i < 8; ++i)
      og[(size_t)(mt * 16 + rbase + i) * CIN + nt * 16 + cc] = acc[i];
  }
}

extern "C" void kernel_launch(void* const* d_in, const int* in_sizes, int n_in,
                              void* d_out, int out_size, void* d_ws, size_t ws_size,
                              hipStream_t stream) {
  (void)in_sizes; (void)n_in; (void)out_size; (void)ws_size;

  const float* x     = (const float*)d_in[0];
  const float* state = (const float*)d_in[1];
  const float* w_o   = (const float*)d_in[14];
  const float* b_o   = (const float*)d_in[15];

  float* out       = (float*)d_out;
  float* state_out = out + (size_t)T_STEPS * BATCH * FREQ * CIN;  // o first, then state
  _Float16* qkv    = (_Float16*)d_ws;  // [3][T][512][64] f16 = 98.3 MB

  gru3_persistent<<<96, 128, 0, stream>>>(
      x, state,
      (const float*)d_in[2],  (const float*)d_in[3],
      (const float*)d_in[4],  (const float*)d_in[5],
      (const float*)d_in[6],  (const float*)d_in[7],
      (const float*)d_in[8],  (const float*)d_in[9],
      (const float*)d_in[10], (const float*)d_in[11],
      (const float*)d_in[12], (const float*)d_in[13],
      qkv, state_out);

  spectral_attention<<<T_STEPS * BATCH, 128, 0, stream>>>(qkv, w_o, b_o, out);
}
